// DiscreteWalker_78632261255281
// MI455X (gfx1250) — compile-verified, matrix operand order checked by reading, not run on hardware
//
#include <hip/hip_runtime.h>
#include <hip/hip_bf16.h>
#include <math.h>

typedef _Float16 f16;
typedef __attribute__((ext_vector_type(16))) _Float16 v16h;
typedef __attribute__((ext_vector_type(8)))  _Float16 v8h;
typedef __attribute__((ext_vector_type(8)))  float    v8f;

#define B_   256
#define TIN_ 128
#define TOUT_ 64
#define IN_  64
#define H_   1024
#define OUT_ 32

// ---------------------------------------------------------------------------
// f16 TN GEMM with WMMA: C[M,N] = A[M,K] * B[N,K]^T (+bias, optional relu)
// A rows K-contiguous (lda), B rows K-contiguous (ldb). fp32 accumulate.
// Optional fp32 output Cf (ldcf) and/or f16 output Ch (ldch).
// Block = 128 threads (4 waves); block tile 64x64; wave tile 32x32 (2x2 WMMA).
// K must be a multiple of 32 (true for all call sites: 32,64,256,1024,1056).
// ---------------------------------------------------------------------------
__device__ __forceinline__ v16h ld_frag(const f16* __restrict__ base, int ld,
                                        int row0, int k0, int lane) {
  // 16-bit A/B fragment: lane L -> row (L&15); halves L>>4 select K sub-block
  // VGPR0-3: k = k0 + (L>>4)*8 + 0..7 ; VGPR4-7: k = k0 + 16 + (L>>4)*8 + 0..7
  const f16* p = base + (long)(row0 + (lane & 15)) * ld + k0 + ((lane >> 4) << 3);
  v8h lo = *(const v8h*)(p);
  v8h hi = *(const v8h*)(p + 16);
  v16h f;
#pragma unroll
  for (int i = 0; i < 8; ++i) { f[i] = lo[i]; f[i + 8] = hi[i]; }
  return f;
}

__global__ __launch_bounds__(128)
void gemm_tn_wmma(const f16* __restrict__ A, const f16* __restrict__ B,
                  const float* __restrict__ bias,
                  float* __restrict__ Cf, f16* __restrict__ Ch,
                  int M, int N, int K,
                  int lda, int ldb, int ldcf, int ldch, int act) {
  const int lane = threadIdx.x & 31;
  const int w    = threadIdx.x >> 5;
  const int m_wave = blockIdx.y * 64 + (w >> 1) * 32;
  const int n_wave = blockIdx.x * 64 + (w & 1) * 32;
  if (m_wave >= M || n_wave >= N) return;   // wave-uniform, EXEC stays full

  v8f acc[2][2] = {};
  for (int k0 = 0; k0 < K; k0 += 32) {
    v16h a0 = ld_frag(A, lda, m_wave,      k0, lane);
    v16h a1 = ld_frag(A, lda, m_wave + 16, k0, lane);
    v16h b0 = ld_frag(B, ldb, n_wave,      k0, lane);
    v16h b1 = ld_frag(B, ldb, n_wave + 16, k0, lane);
    acc[0][0] = __builtin_amdgcn_wmma_f32_16x16x32_f16(false, a0, false, b0, (short)0, acc[0][0], false, false);
    acc[0][1] = __builtin_amdgcn_wmma_f32_16x16x32_f16(false, a0, false, b1, (short)0, acc[0][1], false, false);
    acc[1][0] = __builtin_amdgcn_wmma_f32_16x16x32_f16(false, a1, false, b0, (short)0, acc[1][0], false, false);
    acc[1][1] = __builtin_amdgcn_wmma_f32_16x16x32_f16(false, a1, false, b1, (short)0, acc[1][1], false, false);
  }

  // C/D layout: vgpr v, lane L: row = v + 8*(L>>4), col = (L&15)
  const int rofs = 8 * (lane >> 4);
  const int cofs = lane & 15;
#pragma unroll
  for (int i = 0; i < 2; ++i) {
#pragma unroll
    for (int j = 0; j < 2; ++j) {
      const int col = n_wave + j * 16 + cofs;
      const float bv = bias ? bias[col] : 0.0f;
#pragma unroll
      for (int v = 0; v < 8; ++v) {
        const int row = m_wave + i * 16 + rofs + v;
        float val = acc[i][j][v] + bv;
        if (act) val = fmaxf(val, 0.0f);
        if (Cf) Cf[(long)row * ldcf + col] = val;
        if (Ch) Ch[(long)row * ldch + col] = (f16)val;
      }
    }
  }
}

// ---------------------------------------------------------------------------
// GRU pointwise: gi/gh are [B,3H] (biases already folded in via GEMM epilogue)
// r = sig(gi_r+gh_r); z = sig(gi_z+gh_z); n = tanh(gi_n + r*gh_n)
// h = (1-z)*n + z*h.  Writes fp32 h (in place), f16 h, optional hs slot.
// ---------------------------------------------------------------------------
__global__ void gru_pointwise(const float* __restrict__ gi,
                              const float* __restrict__ gh,
                              float* __restrict__ h,
                              f16* __restrict__ h_h,
                              f16* __restrict__ hs_slot, long hs_row_stride) {
  const int idx = blockIdx.x * blockDim.x + threadIdx.x;
  if (idx >= B_ * H_) return;
  const int b = idx / H_, j = idx - b * H_;
  const long g = (long)b * (3 * H_) + j;
  const float ir = gi[g],          hr = gh[g];
  const float iz = gi[g + H_],     hz = gh[g + H_];
  const float in_ = gi[g + 2*H_],  hn = gh[g + 2*H_];
  const float r = 1.0f / (1.0f + __expf(-(ir + hr)));
  const float z = 1.0f / (1.0f + __expf(-(iz + hz)));
  const float n = tanhf(in_ + r * hn);
  const float ho = h[idx];
  const float hv = (1.0f - z) * n + z * ho;
  h[idx] = hv;
  h_h[idx] = (f16)hv;
  if (hs_slot) hs_slot[(long)b * hs_row_stride + j] = (f16)hv;
}

// scores[b,t] = scale * dot(q[b,:], keys[b,t,:]) ; one wave per (b,t)
__global__ void attn_scores(const float* __restrict__ q,
                            const f16* __restrict__ keys,
                            float* __restrict__ scores, float scale) {
  const int lane = threadIdx.x & 31;
  const int bt = blockIdx.x * (blockDim.x >> 5) + (threadIdx.x >> 5);
  if (bt >= B_ * TIN_) return;
  const int b = bt >> 7;
  const float* qb = q + (long)b * H_;
  const f16*   kb = keys + (long)bt * H_;
  float s = 0.0f;
  for (int hh = lane; hh < H_; hh += 32)
    s += qb[hh] * (float)kb[hh];
#pragma unroll
  for (int off = 16; off > 0; off >>= 1)
    s += __shfl_down(s, off, 32);
  if (lane == 0) scores[bt] = s * scale;
}

// row softmax over TIN_=128 ; one block (128 threads) per batch row
__global__ void softmax128(const float* __restrict__ scores,
                           float* __restrict__ attn) {
  const int b = blockIdx.x, t = threadIdx.x;
  __shared__ float sm[TIN_];
  const float v = scores[b * TIN_ + t];
  sm[t] = v; __syncthreads();
  for (int s = 64; s > 0; s >>= 1) { if (t < s) sm[t] = fmaxf(sm[t], sm[t + s]); __syncthreads(); }
  const float mx = sm[0]; __syncthreads();
  const float e = __expf(v - mx);
  sm[t] = e; __syncthreads();
  for (int s = 64; s > 0; s >>= 1) { if (t < s) sm[t] += sm[t + s]; __syncthreads(); }
  attn[b * TIN_ + t] = e / sm[0];
}

// ctx[b,h] = sum_t attn[b,t]*hs[b,t,h]; write into xcat f16 cols [32, 32+H)
__global__ void ctx_kernel(const float* __restrict__ attn,
                           const f16* __restrict__ hs,
                           f16* __restrict__ xcat) {
  const int b = blockIdx.x;
  const int h = blockIdx.y * blockDim.x + threadIdx.x;
  if (h >= H_) return;
  const float* ab = attn + b * TIN_;
  const f16*   hb = hs + (long)b * TIN_ * H_ + h;
  float s = 0.0f;
  for (int t = 0; t < TIN_; ++t)
    s += ab[t] * (float)hb[(long)t * H_];
  xcat[(long)b * (OUT_ + H_) + OUT_ + h] = (f16)s;
}

__global__ void cvt_f32_to_f16(const float* __restrict__ s, f16* __restrict__ d, int n) {
  const int i = blockIdx.x * blockDim.x + threadIdx.x;
  if (i < n) d[i] = (f16)s[i];
}

__global__ void init_h(const float* __restrict__ h0, float* __restrict__ hf,
                       f16* __restrict__ hh, int n) {
  const int i = blockIdx.x * blockDim.x + threadIdx.x;
  if (i < n) { const float v = h0[i]; hf[i] = v; hh[i] = (f16)v; }
}

// prev0 = x[:, TIN-1, :OUT] -> xcat cols [0,32)
__global__ void init_prev(const float* __restrict__ x, f16* __restrict__ xcat) {
  const int i = blockIdx.x * blockDim.x + threadIdx.x;
  if (i >= B_ * OUT_) return;
  const int b = i / OUT_, o = i - b * OUT_;
  xcat[(long)b * (OUT_ + H_) + o] = (f16)x[(long)b * TIN_ * IN_ + (TIN_ - 1) * IN_ + o];
}

// ---------------------------------------------------------------------------
extern "C" void kernel_launch(void* const* d_in, const int* in_sizes, int n_in,
                              void* d_out, int out_size, void* d_ws, size_t ws_size,
                              hipStream_t stream) {
  const float* x        = (const float*)d_in[0];
  const float* h0_enc   = (const float*)d_in[1];
  const float* h0_dec   = (const float*)d_in[2];
  const float* W_ih_enc = (const float*)d_in[3];
  const float* W_hh_enc = (const float*)d_in[4];
  const float* b_ih_enc = (const float*)d_in[5];
  const float* b_hh_enc = (const float*)d_in[6];
  const float* W_ih_dec = (const float*)d_in[7];
  const float* W_hh_dec = (const float*)d_in[8];
  const float* b_ih_dec = (const float*)d_in[9];
  const float* b_hh_dec = (const float*)d_in[10];
  const float* Wq       = (const float*)d_in[11];
  const float* bq       = (const float*)d_in[12];
  const float* Wk       = (const float*)d_in[13];
  const float* bk       = (const float*)d_in[14];
  const float* W1       = (const float*)d_in[15];
  const float* b1       = (const float*)d_in[16];
  const float* W2       = (const float*)d_in[17];
  const float* b2       = (const float*)d_in[18];
  float* out = (float*)d_out;

  // ---- workspace carve ----
  char* p = (char*)d_ws;
  auto carve = [&](size_t bytes) { void* r = (void*)p; p += (bytes + 255) & ~(size_t)255; return r; };
  f16* x_h        = (f16*)carve((size_t)B_ * TIN_ * IN_ * 2);
  f16* Wih_enc_h  = (f16*)carve((size_t)3 * H_ * IN_ * 2);
  f16* Whh_enc_h  = (f16*)carve((size_t)3 * H_ * H_ * 2);
  f16* Wih_dec_h  = (f16*)carve((size_t)3 * H_ * (OUT_ + H_) * 2);
  f16* Whh_dec_h  = (f16*)carve((size_t)3 * H_ * H_ * 2);
  f16* Wq_h       = (f16*)carve((size_t)H_ * OUT_ * 2);
  f16* Wk_h       = (f16*)carve((size_t)H_ * H_ * 2);
  f16* W1_h       = (f16*)carve((size_t)256 * H_ * 2);
  f16* W2_h       = (f16*)carve((size_t)OUT_ * 256 * 2);
  float* h_enc    = (float*)carve((size_t)B_ * H_ * 4);
  f16*   h_enc_h  = (f16*)carve((size_t)B_ * H_ * 2);
  float* h_dec    = (float*)carve((size_t)B_ * H_ * 4);
  f16*   h_dec_h  = (f16*)carve((size_t)B_ * H_ * 2);
  f16*   hs_h     = (f16*)carve((size_t)B_ * TIN_ * H_ * 2);   // 64 MB (L2-resident)
  f16*   keys_h   = (f16*)carve((size_t)B_ * TIN_ * H_ * 2);   // 64 MB (L2-resident)
  float* gi       = (float*)carve((size_t)B_ * 3 * H_ * 4);
  float* gh       = (float*)carve((size_t)B_ * 3 * H_ * 4);
  float* qbuf     = (float*)carve((size_t)B_ * H_ * 4);
  float* scores   = (float*)carve((size_t)B_ * TIN_ * 4);
  float* attnb    = (float*)carve((size_t)B_ * TIN_ * 4);
  f16*   xcat     = (f16*)carve((size_t)B_ * (OUT_ + H_) * 2);
  f16*   mlp1_h   = (f16*)carve((size_t)B_ * 256 * 2);
  (void)ws_size; (void)n_in; (void)in_sizes; (void)out_size;

  auto cvt = [&](const float* s, f16* d, int n) {
    cvt_f32_to_f16<<<(n + 255) / 256, 256, 0, stream>>>(s, d, n);
  };
  // one-time f16 conversions (re-done every call: deterministic)
  cvt(x, x_h, B_ * TIN_ * IN_);
  cvt(W_ih_enc, Wih_enc_h, 3 * H_ * IN_);
  cvt(W_hh_enc, Whh_enc_h, 3 * H_ * H_);
  cvt(W_ih_dec, Wih_dec_h, 3 * H_ * (OUT_ + H_));
  cvt(W_hh_dec, Whh_dec_h, 3 * H_ * H_);
  cvt(Wq, Wq_h, H_ * OUT_);
  cvt(Wk, Wk_h, H_ * H_);
  cvt(W1, W1_h, 256 * H_);
  cvt(W2, W2_h, OUT_ * 256);
  init_h<<<(B_ * H_ + 255) / 256, 256, 0, stream>>>(h0_enc, h_enc, h_enc_h, B_ * H_);
  init_h<<<(B_ * H_ + 255) / 256, 256, 0, stream>>>(h0_dec, h_dec, h_dec_h, B_ * H_);
  init_prev<<<(B_ * OUT_ + 255) / 256, 256, 0, stream>>>(x, xcat);

  const dim3 gBig((3 * H_) / 64, B_ / 64);     // N=3072, M=256
  const int gruBlocks = (B_ * H_ + 255) / 256;

  // -------- encoder: 128 recurrent steps --------
  for (int t = 0; t < TIN_; ++t) {
    // gi = x_t @ W_ih_enc^T + b_ih_enc   (A rows stride TIN*IN)
    gemm_tn_wmma<<<gBig, 128, 0, stream>>>(x_h + (long)t * IN_, Wih_enc_h, b_ih_enc,
                                           gi, nullptr, B_, 3 * H_, IN_,
                                           TIN_ * IN_, IN_, 3 * H_, 0, 0);
    // gh = h @ W_hh_enc^T + b_hh_enc
    gemm_tn_wmma<<<gBig, 128, 0, stream>>>(h_enc_h, Whh_enc_h, b_hh_enc,
                                           gh, nullptr, B_, 3 * H_, H_,
                                           H_, H_, 3 * H_, 0, 0);
    gru_pointwise<<<gruBlocks, 256, 0, stream>>>(gi, gh, h_enc, h_enc_h,
                                                 hs_h + (long)t * H_, (long)TIN_ * H_);
  }

  // -------- keys = hs @ Wk^T + bk : one big (32768 x 1024 x 1024) GEMM --------
  gemm_tn_wmma<<<dim3(H_ / 64, (B_ * TIN_) / 64), 128, 0, stream>>>(
      hs_h, Wk_h, bk, nullptr, keys_h, B_ * TIN_, H_, H_, H_, H_, 0, H_, 0);

  // -------- decoder: 64 recurrent steps --------
  for (int s = 0; s < TOUT_; ++s) {
    // q = prev @ Wq^T + bq   (prev = xcat cols [0,32), lda = 1056)
    gemm_tn_wmma<<<dim3(H_ / 64, B_ / 64), 128, 0, stream>>>(
        xcat, Wq_h, bq, qbuf, nullptr, B_, H_, OUT_, OUT_ + H_, OUT_, H_, 0, 0);
    attn_scores<<<(B_ * TIN_) / 8, 256, 0, stream>>>(qbuf, keys_h, scores, 1.0f / 32.0f);
    softmax128<<<B_, TIN_, 0, stream>>>(scores, attnb);
    ctx_kernel<<<dim3(B_, H_ / 256), 256, 0, stream>>>(attnb, hs_h, xcat);
    // gi = [prev|ctx] @ W_ih_dec^T + b_ih_dec  (K = 1056)
    gemm_tn_wmma<<<gBig, 128, 0, stream>>>(xcat, Wih_dec_h, b_ih_dec,
                                           gi, nullptr, B_, 3 * H_, OUT_ + H_,
                                           OUT_ + H_, OUT_ + H_, 3 * H_, 0, 0);
    gemm_tn_wmma<<<gBig, 128, 0, stream>>>(h_dec_h, Whh_dec_h, b_hh_dec,
                                           gh, nullptr, B_, 3 * H_, H_,
                                           H_, H_, 3 * H_, 0, 0);
    gru_pointwise<<<gruBlocks, 256, 0, stream>>>(gi, gh, h_dec, h_dec_h, nullptr, 0);
    // mlp1 = relu(h @ W1^T + b1) -> f16
    gemm_tn_wmma<<<dim3(256 / 64, B_ / 64), 128, 0, stream>>>(
        h_dec_h, W1_h, b1, nullptr, mlp1_h, B_, 256, H_, H_, H_, 0, 256, 1);
    // out_s = mlp1 @ W2^T + b2 -> d_out[:, s, :] (f32) AND xcat[:, :32] (f16 prev)
    gemm_tn_wmma<<<dim3(1, B_ / 64), 128, 0, stream>>>(
        mlp1_h, W2_h, b2, out + (long)s * OUT_, xcat, B_, OUT_, 256,
        256, 256, TOUT_ * OUT_, OUT_ + H_, 0);
  }
}
